// Block_11433202942125
// MI455X (gfx1250) — compile-verified, matrix operand order checked by reading, not run on hardware
//
#include <hip/hip_runtime.h>
#include <cstdint>
#include <cstddef>

typedef __attribute__((ext_vector_type(16))) __bf16 v16bf;
typedef __attribute__((ext_vector_type(8)))  __bf16 v8bf;
typedef __attribute__((ext_vector_type(8)))  float  v8f;

#define DEVFN static __device__ __forceinline__

constexpr int kB   = 4;
constexpr int kL   = 2048;
constexpr int kC   = 768;
constexpr int kDS  = 64;
constexpr int kDI  = 1536;
constexpr int kDTR = 48;
constexpr int kNB  = 4;
constexpr int kCB  = 192;
constexpr int kDBLW= kDTR + 2*kDS;   // 176
constexpr int kM   = kB * kL;        // 8192 (multiple of 128: no M guards needed)
constexpr float kEPS   = 1e-6f;
constexpr float kLAMBD = 0.01f;
static_assert(kC == kNB * kCB, "layout");

// ---------------------------------------------------------------- helpers
DEVFN float siluf(float x)      { return x / (1.f + __expf(-x)); }
DEVFN float softplusf(float x)  { return (x > 20.f) ? x : log1pf(__expf(x)); }
DEVFN float sshrinkf(float x)   { return x > kLAMBD ? x - kLAMBD : (x < -kLAMBD ? x + kLAMBD : 0.f); }

// ---------------------------------------------------------------- converts
__global__ void f32_to_bf16_kernel(const float* __restrict__ src, __bf16* __restrict__ dst, size_t n) {
  size_t i = (size_t)blockIdx.x * blockDim.x + threadIdx.x;
  if (i < n) dst[i] = (__bf16)src[i];
}

__global__ void slice_dt_kernel(const float* __restrict__ dbl, __bf16* __restrict__ dst) {
  size_t i = (size_t)blockIdx.x * blockDim.x + threadIdx.x;
  if (i >= (size_t)kM * kDTR) return;
  size_t r = i / kDTR; int c = (int)(i % kDTR);
  dst[i] = (__bf16)dbl[r * kDBLW + c];
}

// emm weight transpose: w (nb,d,k) f32 -> wT (nb,k,d) bf16
__global__ void transpose_w_kernel(const float* __restrict__ w, __bf16* __restrict__ wt) {
  size_t i = (size_t)blockIdx.x * blockDim.x + threadIdx.x;
  if (i >= (size_t)kNB * kCB * kCB) return;
  int d  = (int)(i % kCB);
  int k  = (int)((i / kCB) % kCB);
  int nb = (int)(i / ((size_t)kCB * kCB));
  wt[((size_t)nb * kCB + k) * kCB + d] = (__bf16)w[((size_t)nb * kCB + d) * kCB + k];
}

// ---------------------------------------------------------------- layernorm
__global__ __launch_bounds__(256) void layernorm_kernel(
    const float* __restrict__ x, const float* __restrict__ w, const float* __restrict__ bsh,
    float* __restrict__ outf, __bf16* __restrict__ outb) {
  __shared__ float red[256];
  int row = blockIdx.x, t = threadIdx.x;
  const float* xp = x + (size_t)row * kC;
  float v0 = xp[t], v1 = xp[t + 256], v2 = xp[t + 512];
  red[t] = v0 + v1 + v2;
  __syncthreads();
  for (int o = 128; o > 0; o >>= 1) { if (t < o) red[t] += red[t + o]; __syncthreads(); }
  float mean = red[0] * (1.f / kC);
  __syncthreads();
  float d0 = v0 - mean, d1 = v1 - mean, d2 = v2 - mean;
  red[t] = d0 * d0 + d1 * d1 + d2 * d2;
  __syncthreads();
  for (int o = 128; o > 0; o >>= 1) { if (t < o) red[t] += red[t + o]; __syncthreads(); }
  float inv = rsqrtf(red[0] * (1.f / kC) + kEPS);
  float dd[3] = {d0, d1, d2};
  for (int j = 0; j < 3; ++j) {
    int c = t + j * 256;
    float val = dd[j] * inv * w[c] + bsh[c];
    outf[(size_t)row * kC + c] = val;
    outb[(size_t)row * kC + c] = (__bf16)val;
  }
}

// ---------------------------------------------------------------- WMMA GEMM
constexpr int GBK = 32, GLD = 40;   // K tile, padded LDS row stride (elems)

DEVFN v16bf ldsfrag(const __bf16* base, int row0, int lane) {
  int r = lane & 15, hf = lane >> 4;
  const __bf16* p = base + (size_t)(row0 + r) * GLD + hf * 8;
  v8bf a = *(const v8bf*)p;
  v8bf b = *(const v8bf*)(p + 16);
  v16bf o;
#pragma unroll
  for (int i = 0; i < 8; ++i) { o[i] = a[i]; o[i + 8] = b[i]; }
  return o;
}

DEVFN v16bf negbf(v16bf x) {
#pragma unroll
  for (int i = 0; i < 16; ++i) x[i] = (__bf16)(-(float)x[i]);
  return x;
}

// ---- real GEMM: out[M,N] = A[M,K] * W[N,K]^T. BM=128, BN=128, wave tile 64x32.
// Requires M % 128 == 0, K % 16 == 0. ep: 0 none, 1 +bias, 2 softplus(x+bias)
constexpr int RBM = 128, RBN = 128;
constexpr int RABUF = RBM * GLD, RWBUF = RBN * GLD;

__global__ __launch_bounds__(256) void gemm_bf16_wmma(
    const __bf16* __restrict__ A, int lda, const __bf16* __restrict__ W,
    float* __restrict__ out, int ldo, int N, int K,
    const float* __restrict__ bias, int ep) {
  __shared__ __bf16 As[2 * RABUF];
  __shared__ __bf16 Ws[2 * RWBUF];
  const int tid = threadIdx.x, lane = tid & 31, wave = tid >> 5;
  const int wm = wave >> 2, wn = wave & 3;            // 2x4 wave grid
  const int m0 = blockIdx.y * RBM, n0 = blockIdx.x * RBN;

  // loader: 128 rows x 32 cols, 16 contiguous bf16 per thread (A and W identical shape)
  const int lr = tid >> 1, lch = tid & 1;
  const __bf16* aptr = A + (size_t)(m0 + lr) * lda + lch * 16;
  const int wrow = n0 + lr;
  const bool wok = wrow < N;
  const __bf16* wptr = W + (size_t)wrow * K + lch * 16;
  __bf16* asd = As + (size_t)lr * GLD + lch * 16;
  __bf16* wsd = Ws + (size_t)lr * GLD + lch * 16;

  v8f acc[4][2];
#pragma unroll
  for (int r = 0; r < 4; ++r)
#pragma unroll
    for (int c = 0; c < 2; ++c)
#pragma unroll
      for (int i = 0; i < 8; ++i) acc[r][c][i] = 0.f;

  const uint4 z4 = make_uint4(0u, 0u, 0u, 0u);
  uint4 ra0 = z4, ra1 = z4, rw0 = z4, rw1 = z4;
  const int ktf = K / GBK;

  if (ktf > 0) {  // preload tile 0 into buffer 0 (guard-free on A; W guarded by row only)
    const uint4* pa = (const uint4*)aptr;
    ra0 = pa[0]; ra1 = pa[1];
    rw0 = z4; rw1 = z4;
    if (wok) { const uint4* pw = (const uint4*)wptr; rw0 = pw[0]; rw1 = pw[1]; }
    uint4* qa = (uint4*)asd; qa[0] = ra0; qa[1] = ra1;
    uint4* qw = (uint4*)wsd; qw[0] = rw0; qw[1] = rw1;
  }
  __syncthreads();

  for (int k = 0; k < ktf; ++k) {
    const int cb = k & 1;
    if (k + 1 < ktf) {  // issue next-tile global loads before compute (latency hiding)
      const uint4* pa = (const uint4*)(aptr + (size_t)(k + 1) * GBK);
      ra0 = pa[0]; ra1 = pa[1];
      rw0 = z4; rw1 = z4;
      if (wok) { const uint4* pw = (const uint4*)(wptr + (size_t)(k + 1) * GBK); rw0 = pw[0]; rw1 = pw[1]; }
      if (k + 2 < ktf) {
        __builtin_prefetch(aptr + (size_t)(k + 2) * GBK, 0, 3);
        if (wok) __builtin_prefetch(wptr + (size_t)(k + 2) * GBK, 0, 3);
      }
    }
    const __bf16* ab = As + cb * RABUF;
    const __bf16* wb = Ws + cb * RWBUF;
    v16bf af[4], bfr[2];
#pragma unroll
    for (int r = 0; r < 4; ++r) af[r] = ldsfrag(ab, wm * 64 + r * 16, lane);
#pragma unroll
    for (int c = 0; c < 2; ++c) bfr[c] = ldsfrag(wb, wn * 32 + c * 16, lane);
#pragma unroll
    for (int r = 0; r < 4; ++r)
#pragma unroll
      for (int c = 0; c < 2; ++c)
        acc[r][c] = __builtin_amdgcn_wmma_f32_16x16x32_bf16(false, af[r], false, bfr[c], (short)0, acc[r][c], false, false);
    if (k + 1 < ktf) {
      const int nb2 = cb ^ 1;
      uint4* qa = (uint4*)(asd + nb2 * RABUF); qa[0] = ra0; qa[1] = ra1;
      uint4* qw = (uint4*)(wsd + nb2 * RWBUF); qw[0] = rw0; qw[1] = rw1;
    }
    __syncthreads();
  }

  if (K & (GBK - 1)) {  // single 16-wide K tail (dt_proj K=48); zero-filled upper half
    const int gk = ktf * GBK + lch * 16;
    ra0 = z4; ra1 = z4; rw0 = z4; rw1 = z4;
    if (gk < K) { const uint4* pa = (const uint4*)(aptr + (size_t)ktf * GBK); ra0 = pa[0]; ra1 = pa[1]; }
    if (wok && gk < K) { const uint4* pw = (const uint4*)(wptr + (size_t)ktf * GBK); rw0 = pw[0]; rw1 = pw[1]; }
    uint4* qa = (uint4*)asd; qa[0] = ra0; qa[1] = ra1;
    uint4* qw = (uint4*)wsd; qw[0] = rw0; qw[1] = rw1;
    __syncthreads();
    v16bf af[4], bfr[2];
#pragma unroll
    for (int r = 0; r < 4; ++r) af[r] = ldsfrag(As, wm * 64 + r * 16, lane);
#pragma unroll
    for (int c = 0; c < 2; ++c) bfr[c] = ldsfrag(Ws, wn * 32 + c * 16, lane);
#pragma unroll
    for (int r = 0; r < 4; ++r)
#pragma unroll
      for (int c = 0; c < 2; ++c)
        acc[r][c] = __builtin_amdgcn_wmma_f32_16x16x32_bf16(false, af[r], false, bfr[c], (short)0, acc[r][c], false, false);
  }

  const int col = lane & 15, hf = lane >> 4;
#pragma unroll
  for (int r = 0; r < 4; ++r)
#pragma unroll
    for (int c = 0; c < 2; ++c) {
      const int gc = n0 + wn * 32 + c * 16 + col;
      if (gc >= N) continue;
      const float bv = (bias != nullptr) ? bias[gc] : 0.f;
#pragma unroll
      for (int v = 0; v < 8; ++v) {
        const int gr = m0 + wm * 64 + r * 16 + hf * 8 + v;
        float xv = acc[r][c][v];
        if (ep == 1) xv += bv;
        else if (ep == 2) { xv += bv; xv = softplusf(xv); }
        out[(size_t)gr * ldo + gc] = xv;
      }
    }
}

// ---- complex GEMM for EinFFT. BM=128, BN=64 (N=192 = 3*64 exact), wave tile 32x32.
// Requires M % 128 == 0, K % 32 == 0. ep: 1 relu, 2 softshrink (after bias).
constexpr int CBM = 128, CBN = 64;
constexpr int CABUF = CBM * GLD, CWBUF = CBN * GLD;

__global__ __launch_bounds__(256) void cgemm_bf16_wmma(
    const __bf16* __restrict__ Ar, const __bf16* __restrict__ Ai, int lda, size_t a_z,
    const __bf16* __restrict__ Wr, const __bf16* __restrict__ Wi, size_t w_z,
    const float* __restrict__ br, const float* __restrict__ bi, int b_z,
    float* __restrict__ outR, float* __restrict__ outI, int ldo, size_t o_z,
    __bf16* __restrict__ outRb, __bf16* __restrict__ outIb,
    int N, int K, int ep) {
  __shared__ __bf16 Ars[2 * CABUF], Ais[2 * CABUF];
  __shared__ __bf16 Wrs[2 * CWBUF], Wis[2 * CWBUF];
  const int z = blockIdx.z;
  Ar += (size_t)z * a_z; Ai += (size_t)z * a_z;
  Wr += (size_t)z * w_z; Wi += (size_t)z * w_z;
  const float* brp = br + (size_t)z * b_z;
  const float* bip = bi + (size_t)z * b_z;
  outR += (size_t)z * o_z; outI += (size_t)z * o_z;
  if (outRb) outRb += (size_t)z * o_z;
  if (outIb) outIb += (size_t)z * o_z;

  const int tid = threadIdx.x, lane = tid & 31, wave = tid >> 5;
  const int wm = wave >> 1, wn = wave & 1;
  const int m0 = blockIdx.y * CBM, n0 = blockIdx.x * CBN;

  const int lrA = tid >> 1, lchA = tid & 1;
  const __bf16* arp = Ar + (size_t)(m0 + lrA) * lda + lchA * 16;
  const __bf16* aip = Ai + (size_t)(m0 + lrA) * lda + lchA * 16;
  __bf16* asdR = Ars + (size_t)lrA * GLD + lchA * 16;
  __bf16* asdI = Ais + (size_t)lrA * GLD + lchA * 16;

  const int lrW = tid >> 2, lchW = tid & 3;
  const int wrow = n0 + lrW;
  const bool wok = wrow < N;
  const __bf16* wrp = Wr + (size_t)wrow * K + lchW * 8;
  const __bf16* wip = Wi + (size_t)wrow * K + lchW * 8;
  __bf16* wsdR = Wrs + (size_t)lrW * GLD + lchW * 8;
  __bf16* wsdI = Wis + (size_t)lrW * GLD + lchW * 8;

  v8f aR[2][2], aI[2][2];
#pragma unroll
  for (int r = 0; r < 2; ++r)
#pragma unroll
    for (int c = 0; c < 2; ++c)
#pragma unroll
      for (int i = 0; i < 8; ++i) { aR[r][c][i] = 0.f; aI[r][c][i] = 0.f; }

  const uint4 z4 = make_uint4(0u, 0u, 0u, 0u);
  uint4 r0, r1, i0, i1, wr4, wi4;
  const int ktf = K / GBK;

  {  // preload tile 0
    const uint4* pr = (const uint4*)arp;  r0 = pr[0]; r1 = pr[1];
    const uint4* pi = (const uint4*)aip;  i0 = pi[0]; i1 = pi[1];
    wr4 = z4; wi4 = z4;
    if (wok) { wr4 = *(const uint4*)wrp; wi4 = *(const uint4*)wip; }
    uint4* q;
    q = (uint4*)asdR; q[0] = r0; q[1] = r1;
    q = (uint4*)asdI; q[0] = i0; q[1] = i1;
    *(uint4*)wsdR = wr4; *(uint4*)wsdI = wi4;
  }
  __syncthreads();

  for (int k = 0; k < ktf; ++k) {
    const int cb = k & 1;
    if (k + 1 < ktf) {
      const uint4* pr = (const uint4*)(arp + (size_t)(k + 1) * GBK);  r0 = pr[0]; r1 = pr[1];
      const uint4* pi = (const uint4*)(aip + (size_t)(k + 1) * GBK);  i0 = pi[0]; i1 = pi[1];
      wr4 = z4; wi4 = z4;
      if (wok) {
        wr4 = *(const uint4*)(wrp + (size_t)(k + 1) * GBK);
        wi4 = *(const uint4*)(wip + (size_t)(k + 1) * GBK);
      }
      if (k + 2 < ktf) {
        __builtin_prefetch(arp + (size_t)(k + 2) * GBK, 0, 3);
        __builtin_prefetch(aip + (size_t)(k + 2) * GBK, 0, 3);
      }
    }
    const __bf16* abR = Ars + cb * CABUF;
    const __bf16* abI = Ais + cb * CABUF;
    const __bf16* wbR = Wrs + cb * CWBUF;
    const __bf16* wbI = Wis + cb * CWBUF;
    v16bf arf[2], aif[2], brf[2], bif[2], bnf[2];
#pragma unroll
    for (int r = 0; r < 2; ++r) {
      arf[r] = ldsfrag(abR, wm * 32 + r * 16, lane);
      aif[r] = ldsfrag(abI, wm * 32 + r * 16, lane);
    }
#pragma unroll
    for (int c = 0; c < 2; ++c) {
      brf[c] = ldsfrag(wbR, wn * 32 + c * 16, lane);
      bif[c] = ldsfrag(wbI, wn * 32 + c * 16, lane);
      bnf[c] = negbf(bif[c]);
    }
#pragma unroll
    for (int r = 0; r < 2; ++r)
#pragma unroll
      for (int c = 0; c < 2; ++c) {
        aR[r][c] = __builtin_amdgcn_wmma_f32_16x16x32_bf16(false, arf[r], false, brf[c], (short)0, aR[r][c], false, false);
        aR[r][c] = __builtin_amdgcn_wmma_f32_16x16x32_bf16(false, aif[r], false, bnf[c], (short)0, aR[r][c], false, false);
        aI[r][c] = __builtin_amdgcn_wmma_f32_16x16x32_bf16(false, arf[r], false, bif[c], (short)0, aI[r][c], false, false);
        aI[r][c] = __builtin_amdgcn_wmma_f32_16x16x32_bf16(false, aif[r], false, brf[c], (short)0, aI[r][c], false, false);
      }
    if (k + 1 < ktf) {
      const int nb2 = cb ^ 1;
      uint4* q;
      q = (uint4*)(asdR + nb2 * CABUF); q[0] = r0; q[1] = r1;
      q = (uint4*)(asdI + nb2 * CABUF); q[0] = i0; q[1] = i1;
      *(uint4*)(wsdR + nb2 * CWBUF) = wr4;
      *(uint4*)(wsdI + nb2 * CWBUF) = wi4;
    }
    __syncthreads();
  }

  const int col = lane & 15, hf = lane >> 4;
#pragma unroll
  for (int r = 0; r < 2; ++r)
#pragma unroll
    for (int c = 0; c < 2; ++c) {
      const int gc = n0 + wn * 32 + c * 16 + col;
      if (gc >= N) continue;
      const float bvr = brp[gc], bvi = bip[gc];
#pragma unroll
      for (int v = 0; v < 8; ++v) {
        const int gr = m0 + wm * 32 + r * 16 + hf * 8 + v;
        float xr = aR[r][c][v] + bvr;
        float xi = aI[r][c][v] + bvi;
        if (ep == 1) { xr = fmaxf(xr, 0.f); xi = fmaxf(xi, 0.f); }
        else if (ep == 2) { xr = sshrinkf(xr); xi = sshrinkf(xi); }
        const size_t o = (size_t)gr * ldo + gc;
        outR[o] = xr; outI[o] = xi;
        if (outRb) outRb[o] = (__bf16)xr;
        if (outIb) outIb[o] = (__bf16)xi;
      }
    }
}

// ---------------------------------------------------------------- mamba pieces
__global__ void conv_silu_kernel(const float* __restrict__ xz, const float* __restrict__ cw,
                                 const float* __restrict__ cb, float* __restrict__ xc,
                                 __bf16* __restrict__ xcb) {
  size_t idx = (size_t)blockIdx.x * blockDim.x + threadIdx.x;
  if (idx >= (size_t)kM * kDI) return;
  int d = (int)(idx % kDI);
  size_t bl = idx / kDI;
  int l = (int)(bl % kL);
  size_t b = bl / kL;
  float acc = cb[d];
#pragma unroll
  for (int k = 0; k < 4; ++k) {
    int ls = l + k - 3;
    if (ls >= 0) acc += cw[d * 4 + k] * xz[(b * kL + (size_t)ls) * (2 * kDI) + d];
  }
  float sv = siluf(acc);
  xc[idx] = sv;
  xcb[idx] = (__bf16)sv;
}

__global__ __launch_bounds__(256) void scan_kernel(
    const float* __restrict__ dt, const float* __restrict__ dbl,
    const float* __restrict__ xc, const float* __restrict__ Alog,
    float* __restrict__ ys) {
  int b = blockIdx.y;
  int dl = threadIdx.x >> 2;
  int sp = threadIdx.x & 3;
  int d = blockIdx.x * 64 + dl;
  float Arow[16], h[16];
#pragma unroll
  for (int j = 0; j < 16; ++j) {
    Arow[j] = -__expf(Alog[(size_t)d * kDS + sp * 16 + j]);
    h[j] = 0.f;
  }
  size_t rowbase = (size_t)b * kL;
#pragma unroll 1
  for (int t = 0; t < kL; ++t) {
    size_t row = rowbase + t;
    float dtv = dt[row * kDI + d];
    float u = xc[row * kDI + d];
    const float* bp = dbl + row * kDBLW + kDTR + sp * 16;
    const float* cp = bp + kDS;
    float acc = 0.f;
#pragma unroll
    for (int j = 0; j < 16; ++j) {
      h[j] = h[j] * __expf(dtv * Arow[j]) + dtv * bp[j] * u;
      acc += h[j] * cp[j];
    }
    acc += __shfl_xor(acc, 1, 32);
    acc += __shfl_xor(acc, 2, 32);
    if (sp == 0) ys[row * kDI + d] = acc;
  }
}

__global__ void gate_kernel(const float* __restrict__ ys, const float* __restrict__ xc,
                            const float* __restrict__ xz, const float* __restrict__ Dp,
                            __bf16* __restrict__ yb) {
  size_t idx = (size_t)blockIdx.x * blockDim.x + threadIdx.x;
  if (idx >= (size_t)kM * kDI) return;
  int d = (int)(idx % kDI);
  size_t row = idx / kDI;
  float z = xz[row * (2 * kDI) + kDI + d];
  float y = (ys[idx] + xc[idx] * Dp[d]) * siluf(z);
  yb[idx] = (__bf16)y;
}

__global__ void add_res_kernel(const float* __restrict__ a, const float* __restrict__ b,
                               float* __restrict__ o, size_t n) {
  size_t i = (size_t)blockIdx.x * blockDim.x + threadIdx.x;
  if (i < n) o[i] = a[i] + b[i];
}

// ---------------------------------------------------------------- FFT
// Stockham radix-2 2048-pt FFT over L for one (b,nb,cb); ping-pong LDS, no bit-reversal.
__global__ __launch_bounds__(256) void fftL_fwd_kernel(const float* __restrict__ xin,
                                                       float* __restrict__ outre,
                                                       float* __restrict__ outim) {
  __shared__ float sre[2][2048];
  __shared__ float sim[2][2048];
  int bi = blockIdx.x;
  int cb = bi % kCB;
  int nb = (bi / kCB) % kNB;
  int b = bi / (kCB * kNB);
  size_t base = ((size_t)b * kL * kNB + nb) * kCB + cb;
  int t = threadIdx.x;
  for (int j = 0; j < 8; ++j) {
    int l = t + j * 256;
    sre[0][l] = xin[base + (size_t)l * kC];
    sim[0][l] = 0.f;
  }
  __syncthreads();
  int cur = 0, n = 2048, ls = 0;
  while (n > 1) {
    int m = n >> 1, s = 1 << ls, nxt = cur ^ 1;
    float th = -6.283185307179586f / (float)n;
    for (int j = 0; j < 4; ++j) {
      int idx = t + j * 256;
      int p = idx >> ls, q = idx & (s - 1);
      float ar = sre[cur][q + s * p],        ai = sim[cur][q + s * p];
      float br2 = sre[cur][q + s * (p + m)], bi2 = sim[cur][q + s * (p + m)];
      float sn, cs; __sincosf(th * (float)p, &sn, &cs);
      float dr = ar - br2, di = ai - bi2;
      sre[nxt][q + 2 * s * p] = ar + br2;
      sim[nxt][q + 2 * s * p] = ai + bi2;
      sre[nxt][q + 2 * s * p + s] = dr * cs - di * sn;
      sim[nxt][q + 2 * s * p + s] = dr * sn + di * cs;
    }
    __syncthreads();
    cur = nxt; n = m; ++ls;
  }
  const float sc = 0.022097086912079608f; // 1/sqrt(2048)
  for (int j = 0; j < 8; ++j) {
    int l = t + j * 256;
    outre[base + (size_t)l * kC] = sre[cur][l] * sc;
    outim[base + (size_t)l * kC] = sim[cur][l] * sc;
  }
}

__global__ __launch_bounds__(256) void fftL_inv_kernel(const float* __restrict__ Yre,
                                                       const float* __restrict__ Yim,
                                                       const float* __restrict__ x1,
                                                       float* __restrict__ outp) {
  __shared__ float sre[2][2048];
  __shared__ float sim[2][2048];
  int bi = blockIdx.x;
  int cb = bi % kCB;
  int nb = (bi / kCB) % kNB;
  int b = bi / (kCB * kNB);
  size_t base = ((size_t)b * kL * kNB + nb) * kCB + cb;
  int t = threadIdx.x;
  for (int j = 0; j < 8; ++j) {
    int l = t + j * 256;
    sre[0][l] = Yre[base + (size_t)l * kC];
    sim[0][l] = Yim[base + (size_t)l * kC];
  }
  __syncthreads();
  int cur = 0, n = 2048, ls = 0;
  while (n > 1) {
    int m = n >> 1, s = 1 << ls, nxt = cur ^ 1;
    float th = 6.283185307179586f / (float)n;
    for (int j = 0; j < 4; ++j) {
      int idx = t + j * 256;
      int p = idx >> ls, q = idx & (s - 1);
      float ar = sre[cur][q + s * p],        ai = sim[cur][q + s * p];
      float br2 = sre[cur][q + s * (p + m)], bi2 = sim[cur][q + s * (p + m)];
      float sn, cs; __sincosf(th * (float)p, &sn, &cs);
      float dr = ar - br2, di = ai - bi2;
      sre[nxt][q + 2 * s * p] = ar + br2;
      sim[nxt][q + 2 * s * p] = ai + bi2;
      sre[nxt][q + 2 * s * p + s] = dr * cs - di * sn;
      sim[nxt][q + 2 * s * p + s] = dr * sn + di * cs;
    }
    __syncthreads();
    cur = nxt; n = m; ++ls;
  }
  const float sc = 0.022097086912079608f;
  for (int j = 0; j < 8; ++j) {
    int l = t + j * 256;
    size_t o = base + (size_t)l * kC;
    outp[o] = x1[o] + sre[cur][l] * sc;
  }
}

// 4-point DFT across NB axis (ortho -> *0.5). inv: conj twiddles. Optional fused bf16 outputs.
__global__ void fft4_kernel(float* __restrict__ re, float* __restrict__ im,
                            __bf16* __restrict__ reb, __bf16* __restrict__ imb, int inv) {
  size_t idx = (size_t)blockIdx.x * blockDim.x + threadIdx.x;
  if (idx >= (size_t)kB * kL * kCB) return;
  int cb = (int)(idx % kCB);
  size_t bl = idx / kCB;
  size_t base = bl * (size_t)(kNB * kCB) + cb;
  float r0 = re[base],            i0 = im[base];
  float r1 = re[base + kCB],      i1 = im[base + kCB];
  float r2 = re[base + 2 * kCB],  i2 = im[base + 2 * kCB];
  float r3 = re[base + 3 * kCB],  i3 = im[base + 3 * kCB];
  float s = inv ? 1.f : -1.f;
  float R0 = 0.5f * (r0 + r1 + r2 + r3), I0 = 0.5f * (i0 + i1 + i2 + i3);
  float R2 = 0.5f * (r0 - r1 + r2 - r3), I2 = 0.5f * (i0 - i1 + i2 - i3);
  float R1 = 0.5f * (r0 - s * i1 - r2 + s * i3);
  float I1 = 0.5f * (i0 + s * r1 - i2 - s * r3);
  float R3 = 0.5f * (r0 + s * i1 - r2 - s * i3);
  float I3 = 0.5f * (i0 - s * r1 - i2 + s * r3);
  re[base]           = R0; im[base]           = I0;
  re[base + kCB]     = R1; im[base + kCB]     = I1;
  re[base + 2 * kCB] = R2; im[base + 2 * kCB] = I2;
  re[base + 3 * kCB] = R3; im[base + 3 * kCB] = I3;
  if (reb) {
    reb[base]           = (__bf16)R0; imb[base]           = (__bf16)I0;
    reb[base + kCB]     = (__bf16)R1; imb[base + kCB]     = (__bf16)I1;
    reb[base + 2 * kCB] = (__bf16)R2; imb[base + 2 * kCB] = (__bf16)I2;
    reb[base + 3 * kCB] = (__bf16)R3; imb[base + 3 * kCB] = (__bf16)I3;
  }
}

// ---------------------------------------------------------------- host
static inline int cdiv(long long a, long long b) { return (int)((a + b - 1) / b); }

extern "C" void kernel_launch(void* const* d_in, const int* in_sizes, int n_in,
                              void* d_out, int out_size, void* d_ws, size_t ws_size,
                              hipStream_t stream) {
  (void)in_sizes; (void)n_in; (void)out_size; (void)ws_size;
  const float* x     = (const float*)d_in[0];
  const float* ln1w  = (const float*)d_in[1];
  const float* ln1b  = (const float*)d_in[2];
  const float* ln2w  = (const float*)d_in[3];
  const float* ln2b  = (const float*)d_in[4];
  const float* inpw  = (const float*)d_in[5];
  const float* convw = (const float*)d_in[6];
  const float* convb = (const float*)d_in[7];
  const float* xpw   = (const float*)d_in[8];
  const float* dtw   = (const float*)d_in[9];
  const float* dtb   = (const float*)d_in[10];
  const float* Alog  = (const float*)d_in[11];
  const float* Dv    = (const float*)d_in[12];
  const float* outw  = (const float*)d_in[13];
  const float* wr0   = (const float*)d_in[14];
  const float* wi0   = (const float*)d_in[15];
  const float* br0   = (const float*)d_in[16];
  const float* bi0   = (const float*)d_in[17];
  const float* wr1   = (const float*)d_in[18];
  const float* wi1   = (const float*)d_in[19];
  const float* br1   = (const float*)d_in[20];
  const float* bi1   = (const float*)d_in[21];
  float* outp = (float*)d_out;

  char* ws = (char*)d_ws;
  size_t off = 0;
  auto alloc = [&](size_t bytes) -> void* {
    void* p = ws + off;
    off = (off + bytes + 255) & ~(size_t)255;
    return p;
  };
  __bf16* wInB  = (__bf16*)alloc((size_t)2 * kDI * kC * 2);
  __bf16* wXpB  = (__bf16*)alloc((size_t)kDBLW * kDI * 2);
  __bf16* wDtB  = (__bf16*)alloc((size_t)kDI * kDTR * 2);
  __bf16* wOutB = (__bf16*)alloc((size_t)kC * kDI * 2);
  __bf16* wr0T  = (__bf16*)alloc((size_t)kNB * kCB * kCB * 2);
  __bf16* wi0T  = (__bf16*)alloc((size_t)kNB * kCB * kCB * 2);
  __bf16* wr1T  = (__bf16*)alloc((size_t)kNB * kCB * kCB * 2);
  __bf16* wi1T  = (__bf16*)alloc((size_t)kNB * kCB * kCB * 2);
  __bf16* lnBf  = (__bf16*)alloc((size_t)kM * kC * 2);
  float*  xz    = (float*) alloc((size_t)kM * 2 * kDI * 4);
  float*  xc    = (float*) alloc((size_t)kM * kDI * 4);
  __bf16* xcB   = (__bf16*)alloc((size_t)kM * kDI * 2);
  float*  dblW  = (float*) alloc((size_t)kM * kDBLW * 4);
  __bf16* dtInB = (__bf16*)alloc((size_t)kM * kDTR * 2);
  float*  dtF   = (float*) alloc((size_t)kM * kDI * 4);
  float*  ysW   = (float*) alloc((size_t)kM * kDI * 4);
  __bf16* yB    = (__bf16*)alloc((size_t)kM * kDI * 2);
  float*  mo    = (float*) alloc((size_t)kM * kC * 4);
  float*  x1    = (float*) alloc((size_t)kM * kC * 4);
  float*  ln2f  = (float*) alloc((size_t)kM * kC * 4);
  float*  Xre   = (float*) alloc((size_t)kM * kC * 4);
  float*  Xim   = (float*) alloc((size_t)kM * kC * 4);
  __bf16* XreB  = (__bf16*)alloc((size_t)kM * kC * 2);
  __bf16* XimB  = (__bf16*)alloc((size_t)kM * kC * 2);
  float*  e1r   = (float*) alloc((size_t)kM * kC * 4);
  float*  e1i   = (float*) alloc((size_t)kM * kC * 4);
  __bf16* e1rB  = (__bf16*)alloc((size_t)kM * kC * 2);
  __bf16* e1iB  = (__bf16*)alloc((size_t)kM * kC * 2);
  float*  e2r   = (float*) alloc((size_t)kM * kC * 4);
  float*  e2i   = (float*) alloc((size_t)kM * kC * 4);

  const int T = 256;
  auto f2b = [&](const float* s, __bf16* d, size_t n) {
    f32_to_bf16_kernel<<<cdiv((long long)n, T), T, 0, stream>>>(s, d, n);
  };

  // --- weights to bf16 (recomputed every call; deterministic) ---
  f2b(inpw, wInB, (size_t)2 * kDI * kC);
  f2b(xpw,  wXpB, (size_t)kDBLW * kDI);
  f2b(dtw,  wDtB, (size_t)kDI * kDTR);
  f2b(outw, wOutB,(size_t)kC * kDI);
  {
    int n = kNB * kCB * kCB;
    transpose_w_kernel<<<cdiv(n, T), T, 0, stream>>>(wr0, wr0T);
    transpose_w_kernel<<<cdiv(n, T), T, 0, stream>>>(wi0, wi0T);
    transpose_w_kernel<<<cdiv(n, T), T, 0, stream>>>(wr1, wr1T);
    transpose_w_kernel<<<cdiv(n, T), T, 0, stream>>>(wi1, wi1T);
  }

  // --- Mamba branch ---
  layernorm_kernel<<<kM, 256, 0, stream>>>(x, ln1w, ln1b, ln2f /*scratch*/, lnBf);
  gemm_bf16_wmma<<<dim3(cdiv(2 * kDI, RBN), kM / RBM), 256, 0, stream>>>(
      lnBf, kC, wInB, xz, 2 * kDI, 2 * kDI, kC, nullptr, 0);
  conv_silu_kernel<<<cdiv((long long)kM * kDI, T), T, 0, stream>>>(xz, convw, convb, xc, xcB);
  gemm_bf16_wmma<<<dim3(cdiv(kDBLW, RBN), kM / RBM), 256, 0, stream>>>(
      xcB, kDI, wXpB, dblW, kDBLW, kDBLW, kDI, nullptr, 0);
  slice_dt_kernel<<<cdiv((long long)kM * kDTR, T), T, 0, stream>>>(dblW, dtInB);
  gemm_bf16_wmma<<<dim3(cdiv(kDI, RBN), kM / RBM), 256, 0, stream>>>(
      dtInB, kDTR, wDtB, dtF, kDI, kDI, kDTR, dtb, 2 /*softplus+bias*/);
  scan_kernel<<<dim3(kDI / 64, kB), 256, 0, stream>>>(dtF, dblW, xc, Alog, ysW);
  gate_kernel<<<cdiv((long long)kM * kDI, T), T, 0, stream>>>(ysW, xc, xz, Dv, yB);
  gemm_bf16_wmma<<<dim3(cdiv(kC, RBN), kM / RBM), 256, 0, stream>>>(
      yB, kDI, wOutB, mo, kC, kC, kDI, nullptr, 0);
  add_res_kernel<<<cdiv((long long)kM * kC, T), T, 0, stream>>>(x, mo, x1, (size_t)kM * kC);

  // --- EinFFT branch ---
  layernorm_kernel<<<kM, 256, 0, stream>>>(x1, ln2w, ln2b, ln2f, lnBf);
  fftL_fwd_kernel<<<kB * kNB * kCB, 256, 0, stream>>>(ln2f, Xre, Xim);
  fft4_kernel<<<cdiv((long long)kB * kL * kCB, T), T, 0, stream>>>(Xre, Xim, XreB, XimB, 0);
  cgemm_bf16_wmma<<<dim3(kCB / CBN, kM / CBM, kNB), 256, 0, stream>>>(
      XreB, XimB, kC, (size_t)kCB, wr0T, wi0T, (size_t)kCB * kCB,
      br0, bi0, kCB, e1r, e1i, kC, (size_t)kCB, e1rB, e1iB, kCB, kCB, 1 /*relu*/);
  cgemm_bf16_wmma<<<dim3(kCB / CBN, kM / CBM, kNB), 256, 0, stream>>>(
      e1rB, e1iB, kC, (size_t)kCB, wr1T, wi1T, (size_t)kCB * kCB,
      br1, bi1, kCB, e2r, e2i, kC, (size_t)kCB, nullptr, nullptr, kCB, kCB, 2 /*softshrink*/);
  fft4_kernel<<<cdiv((long long)kB * kL * kCB, T), T, 0, stream>>>(e2r, e2i, nullptr, nullptr, 1);
  fftL_inv_kernel<<<kB * kNB * kCB, 256, 0, stream>>>(e2r, e2i, x1, outp);
}
